// GRNAttentionBias_37314675867809
// MI455X (gfx1250) — compile-verified, matrix operand order checked by reading, not run on hardware
//
#include <hip/hip_runtime.h>
#include <hip/hip_bf16.h>

// GRN-biased attention, fused, CDNA5 (gfx1250) wave32 + WMMA f16->f32.
//
// Per block: one (head, 16-query-row) stripe; 4 waves split the 2048 key
// columns (512 each). Scores stripe in LDS (16 x 2052 f32, padded stride ->
// conflict-free fragment re-reads), exact softmax with exp computed once,
// P*V accumulated on unnormalized exponentials, normalized at the end.
// Attention-weight writes (the 128 MiB bandwidth bottleneck) are b128
// non-temporal stores so K/V/G stay resident in the 192 MB L2.

typedef __attribute__((ext_vector_type(16))) _Float16 v16h;
typedef __attribute__((ext_vector_type(8)))  float    v8f;
typedef __attribute__((ext_vector_type(4)))  float    v4f;

#define SEQ 2048
#define HDIM 64
#define NHEAD 8
#define LDST 2052              // padded LDS row stride in floats (2048+4)
#define NWAVE 4
#define COLS_PER_WAVE (SEQ / NWAVE)   // 512

// Pack a 16x32 WMMA A/B fragment chunk from f32 memory (global or LDS).
// Per-lane element pattern: K = i + (i>=8 ? 8 : 0), i.e. two contiguous
// 8-float runs at base+0 and base+16 -> four b128 loads.
__device__ __forceinline__ v16h pack_frag(const float* __restrict__ base) {
    const v4f* p = (const v4f*)base;
    const v4f a = p[0], b = p[1], c = p[4], d = p[5];
    v16h r;
#pragma unroll
    for (int i = 0; i < 4; ++i) {
        r[i]      = (_Float16)a[i];
        r[i + 4]  = (_Float16)b[i];
        r[i + 8]  = (_Float16)c[i];
        r[i + 12] = (_Float16)d[i];
    }
    return r;
}

__launch_bounds__(128)
__global__ void grn_attn_fused_kernel(const float* __restrict__ Q,
                                      const float* __restrict__ Km,
                                      const float* __restrict__ V,
                                      const float* __restrict__ grn,
                                      const float* __restrict__ hscale,
                                      float* __restrict__ out_o,
                                      float* __restrict__ out_attn)
{
    extern __shared__ float smem[];
    float* sc    = smem;                     // 16 * LDST   score stripe
    float* obuf  = sc + 16 * LDST;           // 16 * 64     O reduction buffer
    float* redm  = obuf + 16 * HDIM;         // NWAVE * 16  per-wave row max
    float* reds  = redm + NWAVE * 16;        // NWAVE * 16  per-wave row sum
    float* gmax  = reds + NWAVE * 16;        // 16          global row max
    float* gisum = gmax + 16;                // 16          1 / global row sum

    const int tid  = threadIdx.x;
    const int wave = tid >> 5;
    const int lane = tid & 31;
    const int half = lane >> 4;              // which 16-lane half of the wave
    const int l16  = lane & 15;
    const int q0   = blockIdx.x * 16;        // query row base
    const int h    = blockIdx.y;             // head
    const float hs = hscale[h];

    const float* Qh = Q  + (size_t)h * SEQ * HDIM;
    const float* Kh = Km + (size_t)h * SEQ * HDIM;
    const float* Vh = V  + (size_t)h * SEQ * HDIM;

    // zero the O reduction buffer
    for (int i = tid; i < 16 * HDIM; i += 128) obuf[i] = 0.0f;

    // ---- Q tile (16 x 64) as two A-fragments (K=0..31, K=32..63) ----
    v16h aq[2];
#pragma unroll
    for (int ch = 0; ch < 2; ++ch)
        aq[ch] = pack_frag(Qh + (size_t)(q0 + l16) * HDIM + ch * 32 + half * 8);

    __syncthreads();

    // ================= Pass A: scores = QK^T/8 + hs*G, track row max ========
    float mpart[8];
#pragma unroll
    for (int j = 0; j < 8; ++j) mpart[j] = -3.0e38f;

    const int cbase = wave * COLS_PER_WAVE;
    for (int t = 0; t < COLS_PER_WAVE / 16; ++t) {
        const int c0 = cbase + t * 16;
        v8f acc = {};
#pragma unroll
        for (int ch = 0; ch < 2; ++ch) {
            // B-fragment: lane = key column N = c0+l16, same K pattern
            const v16h bk =
                pack_frag(Kh + (size_t)(c0 + l16) * HDIM + ch * 32 + half * 8);
            acc = __builtin_amdgcn_wmma_f32_16x16x32_f16(
                false, aq[ch], false, bk, (short)0, acc, false, false);
        }
        // C-layout: element j -> row j+8*half, col = c0 + l16
        const int col = c0 + l16;
#pragma unroll
        for (int j = 0; j < 8; ++j) {
            const int row = j + 8 * half;
            const float s = acc[j] * 0.125f
                          + grn[(size_t)(q0 + row) * SEQ + col] * hs;
            mpart[j] = fmaxf(mpart[j], s);
            sc[row * LDST + col] = s;
        }
    }
    // reduce row max across the 16-lane half (rows live per half)
#pragma unroll
    for (int j = 0; j < 8; ++j) {
#pragma unroll
        for (int m = 1; m < 16; m <<= 1)
            mpart[j] = fmaxf(mpart[j], __shfl_xor(mpart[j], m));
        if (l16 == j) redm[wave * 16 + j + 8 * half] = mpart[j];
    }
    __syncthreads();

    if (tid < 16) {
        float m = redm[tid];
#pragma unroll
        for (int w = 1; w < NWAVE; ++w) m = fmaxf(m, redm[w * 16 + tid]);
        gmax[tid] = m;
    }
    __syncthreads();

    // ================= Pass B1: e = exp(s - max), row sums (vectorized) =====
    // Each wave sweeps its 512-column stripe row-linearly as v4f.
#pragma unroll 1
    for (int row = 0; row < 16; ++row) {
        const float m = gmax[row];
        float acc = 0.0f;
        v4f* srow = (v4f*)(sc + row * LDST + cbase);
#pragma unroll
        for (int it = 0; it < COLS_PER_WAVE / (4 * 32); ++it) {
            const int c4 = it * 32 + lane;
            v4f v = srow[c4];
            v4f e;
#pragma unroll
            for (int k = 0; k < 4; ++k) e[k] = __expf(v[k] - m);
            acc += e[0] + e[1] + e[2] + e[3];
            srow[c4] = e;                  // overwrite with unnormalized weight
        }
#pragma unroll
        for (int m2 = 1; m2 < 32; m2 <<= 1) acc += __shfl_xor(acc, m2);
        if (lane == 0) reds[wave * 16 + row] = acc;
    }

    // ================= Pass B2: P*V on this wave's 512 keys =================
    v8f ao[4] = {v8f{}, v8f{}, v8f{}, v8f{}};
    for (int kt = 0; kt < COLS_PER_WAVE / 32; ++kt) {
        const int k0 = cbase + kt * 32;
        // A-fragment of P from LDS (b128 reads, f32 -> f16)
        const v16h ap = pack_frag(sc + (size_t)l16 * LDST + k0 + half * 8);
#pragma unroll
        for (int nt = 0; nt < 4; ++nt) {
            // B-fragment of V: lane = output dim column d, K = key rows
            v16h bv;
            const int d = nt * 16 + l16;
#pragma unroll
            for (int i = 0; i < 16; ++i) {
                const int kk = i + ((i >= 8) ? 8 : 0) + 8 * half;
                bv[i] = (_Float16)Vh[(size_t)(k0 + kk) * HDIM + d];
            }
            ao[nt] = __builtin_amdgcn_wmma_f32_16x16x32_f16(
                false, ap, false, bv, (short)0, ao[nt], false, false);
        }
    }
    // merge per-wave O partials into LDS (ds_add_f32)
#pragma unroll
    for (int nt = 0; nt < 4; ++nt) {
        const int d = nt * 16 + l16;
#pragma unroll
        for (int j = 0; j < 8; ++j) {
            const int row = j + 8 * half;
            atomicAdd(&obuf[row * HDIM + d], ao[nt][j]);
        }
    }
    __syncthreads();

    if (tid < 16) {
        float s = reds[tid];
#pragma unroll
        for (int w = 1; w < NWAVE; ++w) s += reds[w * 16 + tid];
        gisum[tid] = 1.0f / s;
    }
    __syncthreads();

    // ================= Pass C: write O and attention weights ================
    float* Oout = out_o + ((size_t)h * SEQ + q0) * HDIM;
    {
        const v4f* ob4 = (const v4f*)obuf;
        v4f* od4 = (v4f*)Oout;
        for (int i4 = tid; i4 < 16 * HDIM / 4; i4 += 128) {
            const int row = i4 >> 4;            // 16 v4f per 64-wide row
            v4f v = ob4[i4] * gisum[row];
            od4[i4] = v;
        }
    }

    // Streaming 128 MiB P write: b128 non-temporal stores (keep L2 for K/V/G)
    float* Aout = out_attn + ((size_t)h * SEQ + q0) * SEQ;
#pragma unroll 1
    for (int row = 0; row < 16; ++row) {
        const float inv = gisum[row];
        const v4f* srow = (const v4f*)(sc + row * LDST);
        v4f* drow = (v4f*)(Aout + (size_t)row * SEQ);
        for (int c4 = tid; c4 < SEQ / 4; c4 += 128) {
            v4f v = srow[c4] * inv;
            __builtin_nontemporal_store(v, &drow[c4]);
        }
    }
}

extern "C" void kernel_launch(void* const* d_in, const int* in_sizes, int n_in,
                              void* d_out, int out_size, void* d_ws, size_t ws_size,
                              hipStream_t stream) {
    const float* Q  = (const float*)d_in[0];
    const float* K  = (const float*)d_in[1];
    const float* V  = (const float*)d_in[2];
    const float* G  = (const float*)d_in[3];
    const float* HS = (const float*)d_in[4];

    float* out_o    = (float*)d_out;                       // (1,H,S,D) flat
    float* out_attn = out_o + (size_t)NHEAD * SEQ * HDIM;  // (1,H,S,S) flat

    const size_t shmem =
        (size_t)(16 * LDST + 16 * HDIM + NWAVE * 16 * 2 + 16 + 16) * sizeof(float);

    // allow >64KB dynamic LDS (320KB per WGP on CDNA5)
    static_cast<void>(hipFuncSetAttribute(
        reinterpret_cast<const void*>(&grn_attn_fused_kernel),
        hipFuncAttributeMaxDynamicSharedMemorySize, (int)shmem));

    dim3 grid(SEQ / 16, NHEAD);
    dim3 block(128);
    grn_attn_fused_kernel<<<grid, block, shmem, stream>>>(Q, K, V, G, HS,
                                                          out_o, out_attn);
}
